// SparseAttention_80951543595250
// MI455X (gfx1250) — matrix operand order repaired and checked by reading, not compiled, not run on hardware
//
#include <hip/hip_runtime.h>
#include <hip/hip_bf16.h>

typedef __attribute__((ext_vector_type(16))) _Float16 v16h;
typedef __attribute__((ext_vector_type(8)))  _Float16 v8h;
typedef __attribute__((ext_vector_type(8)))  float    v8f;

#define B_  2
#define T_  2048
#define C_  1024
#define H_  16
#define HD_ 64
#define M_  (B_ * T_)          // 4096 rows
#define XN  (M_ * C_)          // 4,194,304 f16 elements
#define WN  (C_ * C_)          // 1,048,576 f16 elements

__device__ __forceinline__ v8f wmma16(v16h a, v16h b, v8f c) {
    return __builtin_amdgcn_wmma_f32_16x16x32_f16(
        /*neg_a=*/false, a, /*neg_b=*/false, b,
        /*c_mod=*/(short)0, c, /*reuse_a=*/false, /*reuse_b=*/false);
}

// A operand (16-bit 16x32 tile): lane&15 = M row; half = lane>>4:
//   halves 0..15 -> K = 16*half + (0..15) (this code's order; B loads match)
__device__ __forceinline__ v16h load_a(const _Float16* p) {
    v8h lo = *(const v8h*)(p);
    v8h hi = *(const v8h*)(p + 8);
    v16h a;
#pragma unroll
    for (int i = 0; i < 8; ++i) { a[i] = lo[i]; a[i + 8] = hi[i]; }
    return a;
}

// ---------------------------------------------------------------- convert
__global__ __launch_bounds__(256) void f32_to_f16_kernel(
    const float* __restrict__ src, _Float16* __restrict__ dst, int n) {
    int i = blockIdx.x * blockDim.x + threadIdx.x;
    if (i < n) dst[i] = (_Float16)src[i];
}

// ---------------------------------------------------------------- fused QKV
// {Q,K,V}[M,N] = X[M,K] @ W{q,k,v}[N,K]^T + b{q,k,v}
// One wave per 32x64 output block, shared across the 3 projections:
// 2 A loads feed 24 WMMAs per K-step.
__global__ __launch_bounds__(32) void qkv_gemm_kernel(
    const _Float16* __restrict__ X,
    const _Float16* __restrict__ Wq, const _Float16* __restrict__ Wk,
    const _Float16* __restrict__ Wv,
    const float* __restrict__ bq, const float* __restrict__ bk,
    const float* __restrict__ bv,
    _Float16* __restrict__ Q, _Float16* __restrict__ Ko,
    _Float16* __restrict__ V) {
    constexpr int K = C_, N = C_;
    const int lane = threadIdx.x & 31;
    const int col  = lane & 15;
    const int half = lane >> 4;
    const int n0 = blockIdx.x * 64;
    const int m0 = blockIdx.y * 32;

    const _Float16* Ws[3] = {Wq, Wk, Wv};
    const float*    bs[3] = {bq, bk, bv};
    _Float16*       Os[3] = {Q, Ko, V};

    const _Float16* xp0 = X + (size_t)(m0 + col) * K + 16 * half;
    const _Float16* xp1 = xp0 + (size_t)16 * K;
    const _Float16* wp[3][4];
#pragma unroll
    for (int p = 0; p < 3; ++p)
#pragma unroll
        for (int j = 0; j < 4; ++j)
            wp[p][j] = Ws[p] + (size_t)(n0 + 16 * j + col) * K + 16 * half;

    v8f acc[3][2][4] = {};
    for (int k0 = 0; k0 < K; k0 += 32) {
        __builtin_prefetch(xp0 + k0 + 256, 0, 0);
        __builtin_prefetch(xp1 + k0 + 256, 0, 0);
        v16h a0 = load_a(xp0 + k0);
        v16h a1 = load_a(xp1 + k0);
#pragma unroll
        for (int p = 0; p < 3; ++p) {
#pragma unroll
            for (int j = 0; j < 4; ++j) {
                v16h b = *(const v16h*)(wp[p][j] + k0);
                acc[p][0][j] = wmma16(a0, b, acc[p][0][j]);
                acc[p][1][j] = wmma16(a1, b, acc[p][1][j]);
            }
        }
    }

    // C/D layout: vgpr r holds M = r + 8*half, N = col (per 16x16 tile)
#pragma unroll
    for (int p = 0; p < 3; ++p) {
#pragma unroll
        for (int j = 0; j < 4; ++j) {
            const int n = n0 + 16 * j + col;
            const float bv_ = bs[p][n];
#pragma unroll
            for (int mi = 0; mi < 2; ++mi)
#pragma unroll
                for (int r = 0; r < 8; ++r) {
                    int m = m0 + 16 * mi + r + 8 * half;
                    Os[p][(size_t)m * N + n] = (_Float16)(acc[p][mi][j][r] + bv_);
                }
        }
    }
}

// ---------------------------------------------------------------- GEMM (32x64 block)
template<bool OUTF16>
__global__ __launch_bounds__(32) void gemm_bias_kernel(
    const _Float16* __restrict__ X, const _Float16* __restrict__ W,
    const float* __restrict__ bias, _Float16* __restrict__ Yh,
    float* __restrict__ Yf, int M, int N, int K) {
    const int lane = threadIdx.x & 31;
    const int col  = lane & 15;
    const int half = lane >> 4;
    const int n0 = blockIdx.x * 64;
    const int m0 = blockIdx.y * 32;

    const _Float16* xp0 = X + (size_t)(m0 + col) * K + 16 * half;
    const _Float16* xp1 = xp0 + (size_t)16 * K;
    const _Float16* wp[4];
#pragma unroll
    for (int j = 0; j < 4; ++j)
        wp[j] = W + (size_t)(n0 + 16 * j + col) * K + 16 * half;

    v8f acc[2][4] = {};
    for (int k0 = 0; k0 < K; k0 += 32) {
        __builtin_prefetch(xp0 + k0 + 256, 0, 0);
        __builtin_prefetch(xp1 + k0 + 256, 0, 0);
        v16h a0 = load_a(xp0 + k0);
        v16h a1 = load_a(xp1 + k0);
#pragma unroll
        for (int j = 0; j < 4; ++j) {
            __builtin_prefetch(wp[j] + k0 + 256, 0, 0);
            v16h b = *(const v16h*)(wp[j] + k0);
            acc[0][j] = wmma16(a0, b, acc[0][j]);
            acc[1][j] = wmma16(a1, b, acc[1][j]);
        }
    }

#pragma unroll
    for (int j = 0; j < 4; ++j) {
        const int n = n0 + 16 * j + col;
        const float bv = bias[n];
#pragma unroll
        for (int mi = 0; mi < 2; ++mi)
#pragma unroll
            for (int r = 0; r < 8; ++r) {
                int m = m0 + 16 * mi + r + 8 * half;
                float y = acc[mi][j][r] + bv;
                if (OUTF16) Yh[(size_t)m * N + n] = (_Float16)y;
                else        Yf[(size_t)m * N + n] = y;
            }
    }
}

// ---------------------------------------------------------------- attention
// One wave per (b, h, 16-query tile). Window: j in [i-32, i+32).
// Key union for a 16-row tile = 96 keys (6 key tiles of 16).
__global__ __launch_bounds__(32) void attn_kernel(
    const _Float16* __restrict__ qh, const _Float16* __restrict__ kh,
    const _Float16* __restrict__ vh, _Float16* __restrict__ ah) {
    const int lane = threadIdx.x & 31;
    const int col  = lane & 15;
    const int half = lane >> 4;

    const int qt = blockIdx.x & 127;           // T/16 tiles
    const int h  = (blockIdx.x >> 7) & 15;
    const int b  = blockIdx.x >> 11;
    const int m0 = qt * 16;
    const size_t bT = (size_t)b * T_;

    const int LDST = 104;                      // padded row stride (halves)
    __shared__ _Float16 sAttn[16 * 104];

    // ---- Q tile as two A operands (16 x 64 over K-steps 0,32)
    v16h a0, a1;
    {
        const _Float16* qp = qh + (bT + m0 + col) * (size_t)C_ + h * HD_ + 16 * half;
        a0 = load_a(qp);
        a1 = load_a(qp + 32);
    }

    // ---- scores: 6 key tiles, S = q @ k^T  (B column n = k row of key n)
    const int kb0 = m0 - 32;
    v8f S[6];
#pragma unroll
    for (int t = 0; t < 6; ++t) {
        int key = kb0 + t * 16 + col;
        int kc  = key < 0 ? 0 : (key > T_ - 1 ? T_ - 1 : key);
        const _Float16* kp = kh + (bT + kc) * (size_t)C_ + h * HD_ + 16 * half;
        v16h b0 = *(const v16h*)(kp);
        v16h b1 = *(const v16h*)(kp + 32);
        v8f acc = {};
        acc = wmma16(a0, b0, acc);
        acc = wmma16(a1, b1, acc);
        S[t] = acc;
    }

    // ---- mask + softmax (row m0+r+8*half lives in vgpr r across 16 lanes)
    const float scale = 0.125f;                // HD^-0.5
#pragma unroll
    for (int r = 0; r < 8; ++r) {
        const int i = m0 + r + 8 * half;
        float mx = -1e30f;
#pragma unroll
        for (int t = 0; t < 6; ++t) {
            int j = kb0 + t * 16 + col;
            bool ok = (j >= i - 32) && (j < i + 32) && (j >= 0) && (j < T_);
            float s = ok ? S[t][r] * scale : -1e30f;
            S[t][r] = s;
            mx = fmaxf(mx, s);
        }
#pragma unroll
        for (int o = 1; o < 16; o <<= 1) mx = fmaxf(mx, __shfl_xor(mx, o, 32));
        float sum = 0.f;
#pragma unroll
        for (int t = 0; t < 6; ++t) {
            float e = __expf(S[t][r] - mx);
            S[t][r] = e;
            sum += e;
        }
#pragma unroll
        for (int o = 1; o < 16; o <<= 1) sum += __shfl_xor(sum, o, 32);
        float inv = 1.f / sum;
#pragma unroll
        for (int t = 0; t < 6; ++t)
            sAttn[(r + 8 * half) * LDST + t * 16 + col] = (_Float16)(S[t][r] * inv);
    }
    __syncthreads();

    // ---- A operands for attn @ V (K = 96, three K-steps of 32) from LDS
    v16h A2[3];
#pragma unroll
    for (int s = 0; s < 3; ++s)
        A2[s] = load_a(&sAttn[col * LDST + 32 * s + 16 * half]);

    // ---- out = attn @ V, N = head-dim in 4 tiles of 16
#pragma unroll
    for (int d0 = 0; d0 < HD_; d0 += 16) {
        v8f acc = {};
#pragma unroll
        for (int s = 0; s < 3; ++s) {
            v16h bm;
#pragma unroll
            for (int j = 0; j < 16; ++j) {
                int key = kb0 + 32 * s + 16 * half + j;
                key = key < 0 ? 0 : (key > T_ - 1 ? T_ - 1 : key);
                bm[j] = vh[(bT + key) * (size_t)C_ + h * HD_ + d0 + col];
            }
            acc = wmma16(A2[s], bm, acc);
        }
#pragma unroll
        for (int r = 0; r < 8; ++r) {
            int m = m0 + r + 8 * half;
            ah[(bT + m) * (size_t)C_ + h * HD_ + d0 + col] = (_Float16)acc[r];
        }
    }
}

// ---------------------------------------------------------------- launch
extern "C" void kernel_launch(void* const* d_in, const int* in_sizes, int n_in,
                              void* d_out, int out_size, void* d_ws, size_t ws_size,
                              hipStream_t stream) {
    const float* x  = (const float*)d_in[0];
    const float* qw = (const float*)d_in[1];
    const float* qb = (const float*)d_in[2];
    const float* kw = (const float*)d_in[3];
    const float* kb = (const float*)d_in[4];
    const float* vw = (const float*)d_in[5];
    const float* vb = (const float*)d_in[6];
    const float* ow = (const float*)d_in[7];
    const float* ob = (const float*)d_in[8];
    float* out = (float*)d_out;

    _Float16* ws = (_Float16*)d_ws;
    _Float16* xh  = ws;                 // [M_, C_] f16 x   (reused as attn-out)
    _Float16* qwh = xh  + XN;           // [C_, C_]
    _Float16* kwh = qwh + WN;
    _Float16* vwh = kwh + WN;
    _Float16* owh = vwh + WN;
    _Float16* qhh = owh + WN;           // [M_, C_] q
    _Float16* khh = qhh + XN;           // k
    _Float16* vhh = khh + XN;           // v
    _Float16* ahh = xh;                 // attn out aliases xh (x dead by then)

    // 1) f32 -> f16 conversions
    f32_to_f16_kernel<<<(XN + 255) / 256, 256, 0, stream>>>(x,  xh,  XN);
    f32_to_f16_kernel<<<(WN + 255) / 256, 256, 0, stream>>>(qw, qwh, WN);
    f32_to_f16_kernel<<<(WN + 255) / 256, 256, 0, stream>>>(kw, kwh, WN);
    f32_to_f16_kernel<<<(WN + 255) / 256, 256, 0, stream>>>(vw, vwh, WN);
    f32_to_f16_kernel<<<(WN + 255) / 256, 256, 0, stream>>>(ow, owh, WN);

    // 2) fused QKV projection (WMMA, 32x64 blocks, shared A operand)
    dim3 qgrid(C_ / 64, M_ / 32);
    qkv_gemm_kernel<<<qgrid, 32, 0, stream>>>(xh, qwh, kwh, vwh,
                                              qb, kb, vb, qhh, khh, vhh);

    // 3) windowed attention (WMMA scores + WMMA attn@V)
    attn_kernel<<<B_ * H_ * (T_ / 16), 32, 0, stream>>>(qhh, khh, vhh, ahh);

    // 4) output projection (WMMA, 32x64 blocks), f32 into d_out
    gemm_bias_kernel<false><<<qgrid, 32, 0, stream>>>(ahh, owh, ob, nullptr, out,
                                                      M_, C_, C_);
}
